// X_Attention_E2E_LD_75471165325440
// MI455X (gfx1250) — matrix operand order repaired and checked by reading, not compiled, not run on hardware
//
#include <hip/hip_runtime.h>

typedef __bf16 bf16;
typedef __attribute__((ext_vector_type(16))) __bf16 v16bf;
typedef __attribute__((ext_vector_type(8)))  float  v8f;

#define NGROUP 17920   // B*T = 128*140
#define NTOK   140
#define NBATCH 128

// ---------------------------------------------------------------------------
// CDNA5 async Global->LDS copies, tracked by ASYNCcnt (ISA 15.18.3 op 96/98).
// VDST = VGPR with LDS byte offset (low 32 bits of generic LDS address),
// VADDR = 64-bit global address, GV mode (saddr = off).
// ---------------------------------------------------------------------------
__device__ __forceinline__ void async_copy_b128(void* lds_dst, const void* gsrc)
{
    unsigned lds_off = (unsigned)(size_t)lds_dst;
    unsigned long long ga = (unsigned long long)(size_t)gsrc;
    asm volatile("global_load_async_to_lds_b128 %0, %1, off"
                 :: "v"(lds_off), "v"(ga) : "memory");
}
__device__ __forceinline__ void async_copy_b32(void* lds_dst, const void* gsrc)
{
    unsigned lds_off = (unsigned)(size_t)lds_dst;
    unsigned long long ga = (unsigned long long)(size_t)gsrc;
    asm volatile("global_load_async_to_lds_b32 %0, %1, off"
                 :: "v"(lds_off), "v"(ga) : "memory");
}
// wait until at most N async ops remain in flight (async loads retire in order)
template<int N>
__device__ __forceinline__ void async_wait_le()
{
    asm volatile("s_wait_asynccnt %0" :: "i"(N) : "memory");
}

// ---------------------------------------------------------------------------
// Stage-1: per-chunk frame MLP (99->99->99) + chunk softmax pooling -> new_x
// bf16 [NGROUP x 128] (cols 99..127 zero so GEMMs can run padded K=128).
// ---------------------------------------------------------------------------
__global__ void frame_mlp_kernel(const float* __restrict__ x,
                                 const float* __restrict__ w1, const float* __restrict__ b1,
                                 const float* __restrict__ w2, const float* __restrict__ b2,
                                 const float* __restrict__ w3, const float* __restrict__ b3,
                                 bf16* __restrict__ newx)
{
    __shared__ float xs[10 * 99];
    __shared__ float h1[10 * 99];
    __shared__ float h2[10 * 99];
    __shared__ float wts[10];
    int g = blockIdx.x, tid = threadIdx.x;            // 128 threads
    const float* xb = x + (size_t)g * 10 * 99;
    for (int i = tid; i < 990; i += 128) xs[i] = xb[i];
    __syncthreads();
    for (int i = tid; i < 990; i += 128) {
        int f = i / 99, j = i - f * 99;
        float s = b1[j];
        for (int k = 0; k < 99; ++k) s += xs[f * 99 + k] * w1[k * 99 + j];
        h1[i] = fmaxf(s, 0.f);
    }
    __syncthreads();
    for (int i = tid; i < 990; i += 128) {
        int f = i / 99, j = i - f * 99;
        float s = b2[j];
        for (int k = 0; k < 99; ++k) s += h1[f * 99 + k] * w2[k * 99 + j];
        h2[i] = fmaxf(s, 0.f);
    }
    __syncthreads();
    if (tid < 10) {
        float s = b3[0];
        for (int k = 0; k < 99; ++k) s += h2[tid * 99 + k] * w3[k];
        wts[tid] = fmaxf(s, 0.f);
    }
    __syncthreads();
    if (tid == 0) {
        float m = wts[0];
        for (int f = 1; f < 10; ++f) m = fmaxf(m, wts[f]);
        float su = 0.f;
        for (int f = 0; f < 10; ++f) { float e = __expf(wts[f] - m); wts[f] = e; su += e; }
        float inv = 1.f / su;
        for (int f = 0; f < 10; ++f) wts[f] *= inv;
    }
    __syncthreads();
    {
        int j = tid;   // 128 threads, ld = 128 (pad with zeros)
        float s = 0.f;
        if (j < 99)
            for (int f = 0; f < 10; ++f) s += h2[f * 99 + j] * wts[f];
        newx[(size_t)g * 128 + j] = (bf16)s;
    }
}

// ---------------------------------------------------------------------------
// Weight transpose+convert: fp32 [K x N] row-major -> bf16 [N x Kp] (zero pad
// K..Kp). One-time per launch; makes GEMM B-panels contiguous 16B chunks.
// ---------------------------------------------------------------------------
__global__ void wt_conv_kernel(const float* __restrict__ W, int K, int N, int Kp,
                               bf16* __restrict__ Wt)
{
    __shared__ float t[32][33];
    int k0 = blockIdx.x * 32, n0 = blockIdx.y * 32;
    int tx = threadIdx.x & 31, ty = threadIdx.x >> 5;   // 32 x 8
    for (int r = ty; r < 32; r += 8) {
        int k = k0 + r, n = n0 + tx;
        t[r][tx] = (k < K && n < N) ? W[(size_t)k * N + n] : 0.f;
    }
    __syncthreads();
    for (int r = ty; r < 32; r += 8) {
        int n = n0 + r;
        if (n < N) Wt[(size_t)n * Kp + k0 + tx] = (bf16)t[tx][r];
    }
}

// in-place BatchNorm apply on a bf16 activation: a = a*scale[c] + shift[c]
__global__ void bn_apply_kernel(bf16* __restrict__ A, int M, int N, int ld,
                                const float* __restrict__ sc, const float* __restrict__ sh)
{
    size_t i = (size_t)blockIdx.x * 256 + threadIdx.x;
    if (i >= (size_t)M * N) return;
    size_t r = i / (size_t)N;
    int c = (int)(i - r * (size_t)N);
    float v = (float)A[r * ld + c];
    A[r * ld + c] = (bf16)(v * sc[c] + sh[c]);
}

// ---------------------------------------------------------------------------
// bf16 WMMA GEMM: one wave per 16x64 output tile (4 accumulators, one shared
// A fragment). Double-buffered LDS; tiles streamed with async Global->LDS
// b128 copies pipelined against the WMMAs via partial ASYNCcnt waits.
// ---------------------------------------------------------------------------
__global__ void gemm_bf16_kernel(const bf16* __restrict__ A, int lda,
                                 const bf16* __restrict__ Wt, int Kp, int N,
                                 const float* __restrict__ bias,
                                 float* __restrict__ out_f32, int ldc_f, int relu_f,
                                 bf16* __restrict__ out_h, int ldc_h, int relu_h)
{
    __shared__ bf16 As[2][16 * 32];     // [m][k]
    __shared__ bf16 Bs[2][64 * 32];     // B^T [n][k]
    int row0 = blockIdx.x * 16;
    int col0 = blockIdx.y * 64;
    int lane = threadIdx.x;             // wave32
    int m = lane & 15, half = lane >> 4;
    v8f acc[4] = {};

    auto stage = [&](int buf, int k0) {   // 10 async ops per lane (uniform)
        #pragma unroll
        for (int s = 0; s < 2; ++s) {            // A tile: 64 x 16B slots
            int slot = lane + s * 32;
            int r = slot >> 2, q = slot & 3;
            async_copy_b128(&As[buf][r * 32 + q * 8],
                            &A[(size_t)(row0 + r) * lda + k0 + q * 8]);
        }
        #pragma unroll
        for (int s = 0; s < 8; ++s) {            // B tile: 256 x 16B slots
            int slot = lane + s * 32;
            int n = slot >> 2, q = slot & 3;
            async_copy_b128(&Bs[buf][n * 32 + q * 8],
                            &Wt[(size_t)(col0 + n) * Kp + k0 + q * 8]);
        }
    };

    stage(0, 0);
    for (int k0 = 0; k0 < Kp; k0 += 32) {
        int buf = (k0 >> 5) & 1;
        bool have_next = (k0 + 32) < Kp;
        if (lane == 0 && k0 + 64 < Kp) {         // L2 prefetch 2 stages ahead
            __builtin_prefetch(&A[(size_t)row0 * lda + k0 + 64], 0, 1);
            __builtin_prefetch(&Wt[(size_t)col0 * Kp + k0 + 64], 0, 1);
        }
        if (have_next) { stage(buf ^ 1, k0 + 32); async_wait_le<10>(); }
        else           { async_wait_le<0>(); }
        __syncthreads();
        v16bf af;
        const bf16* ap = &As[buf][(m << 5) + half * 8];
        #pragma unroll
        for (int j = 0; j < 8; ++j) { af[j] = ap[j]; af[8 + j] = ap[16 + j]; }
        #pragma unroll
        for (int s = 0; s < 4; ++s) {
            v16bf bv;
            const bf16* bp = &Bs[buf][((s * 16 + m) << 5) + half * 8];
            #pragma unroll
            for (int j = 0; j < 8; ++j) { bv[j] = bp[j]; bv[8 + j] = bp[16 + j]; }
            acc[s] = __builtin_amdgcn_wmma_f32_16x16x32_bf16(false, af, false, bv,
                                                             (short)0, acc[s], false, false);
        }
        __syncthreads();    // buf may be restaged next iteration
    }
    #pragma unroll
    for (int s = 0; s < 4; ++s) {
        int gcol = col0 + s * 16 + m;
        float bvv = (bias && gcol < N) ? bias[gcol] : 0.f;
        #pragma unroll
        for (int r = 0; r < 8; ++r) {
            int grow = row0 + half * 8 + r;
            float v = acc[s][r] + bvv;
            if (out_f32 && gcol < N)
                out_f32[(size_t)grow * ldc_f + gcol] = relu_f ? fmaxf(v, 0.f) : v;
            if (out_h && gcol < ldc_h) {
                float vb = relu_h ? fmaxf(v, 0.f) : v;
                out_h[(size_t)grow * ldc_h + gcol] = (gcol < N) ? (bf16)vb : (bf16)0.f;
            }
        }
    }
}

// ---------------------------------------------------------------------------
// WMMA GEMM (N fixed = 256) + fused residual + LayerNorm epilogue.
// Block = 256 threads (8 waves); wave w owns columns [w*32, w*32+32).
// Double-buffered async staging; every wave issues exactly 5 async ops per
// stage (A via one b32/thread, B via 4 b128/thread) so the partial
// s_wait_asynccnt immediate is uniform across waves.
// ---------------------------------------------------------------------------
__global__ void gemm_ln_kernel(const bf16* __restrict__ A, int lda, int Kp,
                               const bf16* __restrict__ Wt,            // [256 x Kp]
                               const float* __restrict__ bias,
                               const float* __restrict__ res,          // [M x 256] f32
                               const float* __restrict__ gamma,
                               const float* __restrict__ beta,
                               float* __restrict__ of, bf16* __restrict__ oh)
{
    __shared__ bf16  As[2][16 * 32];
    __shared__ bf16  Bt[2][256 * 32];
    __shared__ float Ds[16 * 256];
    __shared__ float redA[16 * 17], redB[16 * 17];
    __shared__ float mrow[16], rrow[16];
    int row0 = blockIdx.x * 16;
    int tid = threadIdx.x;
    int wave = tid >> 5, lane = tid & 31;
    int m = lane & 15, half = lane >> 4;
    v8f acc0 = {}, acc1 = {};

    auto stage = [&](int buf, int k0) {   // 5 async ops per wave (uniform)
        {                                          // A tile: 256 x 4B slots
            int r = tid >> 4, q = tid & 15;
            async_copy_b32(&As[buf][r * 32 + q * 2],
                           &A[(size_t)(row0 + r) * lda + k0 + q * 2]);
        }
        #pragma unroll
        for (int s = 0; s < 4; ++s) {              // B tile: 1024 x 16B slots
            int slot = tid + s * 256;
            int n = slot >> 2, q = slot & 3;
            async_copy_b128(&Bt[buf][n * 32 + q * 8],
                            &Wt[(size_t)n * Kp + k0 + q * 8]);
        }
    };

    stage(0, 0);
    for (int k0 = 0; k0 < Kp; k0 += 32) {
        int buf = (k0 >> 5) & 1;
        bool have_next = (k0 + 32) < Kp;
        if (have_next) { stage(buf ^ 1, k0 + 32); async_wait_le<5>(); }
        else           { async_wait_le<0>(); }
        __syncthreads();
        v16bf af;
        const bf16* ap = &As[buf][(m << 5) + half * 8];
        #pragma unroll
        for (int j = 0; j < 8; ++j) { af[j] = ap[j]; af[8 + j] = ap[16 + j]; }
        {
            v16bf bv; const bf16* bp = &Bt[buf][((wave * 32 + m) << 5) + half * 8];
            #pragma unroll
            for (int j = 0; j < 8; ++j) { bv[j] = bp[j]; bv[8 + j] = bp[16 + j]; }
            acc0 = __builtin_amdgcn_wmma_f32_16x16x32_bf16(false, af, false, bv,
                                                           (short)0, acc0, false, false);
        }
        {
            v16bf bv; const bf16* bp = &Bt[buf][((wave * 32 + 16 + m) << 5) + half * 8];
            #pragma unroll
            for (int j = 0; j < 8; ++j) { bv[j] = bp[j]; bv[8 + j] = bp[16 + j]; }
            acc1 = __builtin_amdgcn_wmma_f32_16x16x32_bf16(false, af, false, bv,
                                                           (short)0, acc1, false, false);
        }
        __syncthreads();
    }
    #pragma unroll
    for (int r = 0; r < 8; ++r) {
        Ds[(half * 8 + r) * 256 + wave * 32 + m]      = acc0[r];
        Ds[(half * 8 + r) * 256 + wave * 32 + 16 + m] = acc1[r];
    }
    __syncthreads();
    int row = tid >> 4, j = tid & 15;
    float s = 0.f, s2 = 0.f;
    for (int c = j; c < 256; c += 16) {
        float v = Ds[row * 256 + c] + bias[c] + res[(size_t)(row0 + row) * 256 + c];
        Ds[row * 256 + c] = v;
        s += v; s2 += v * v;
    }
    redA[row * 17 + j] = s; redB[row * 17 + j] = s2;
    __syncthreads();
    if (j == 0) {
        float ts = 0.f, ts2 = 0.f;
        for (int k = 0; k < 16; ++k) { ts += redA[row * 17 + k]; ts2 += redB[row * 17 + k]; }
        float mean = ts / 256.f;
        float var  = ts2 / 256.f - mean * mean;
        mrow[row] = mean; rrow[row] = rsqrtf(var + 1e-6f);
    }
    __syncthreads();
    for (int c = j; c < 256; c += 16) {
        float v = Ds[row * 256 + c];
        float o = gamma[c] * (v - mrow[row]) * rrow[row] + beta[c];
        of[(size_t)(row0 + row) * 256 + c] = o;
        oh[(size_t)(row0 + row) * 256 + c] = (bf16)o;
    }
}

// ---------------------------------------------------------------------------
// Column mean/var over M rows of a bf16 activation -> BN scale/shift
// ---------------------------------------------------------------------------
__global__ void col_stats_kernel(const bf16* __restrict__ A, int M, int ld, float eps,
                                 float* __restrict__ scale, float* __restrict__ shift)
{
    int c = blockIdx.x, tid = threadIdx.x;
    __shared__ float s1[256], s2[256];
    float a = 0.f, b = 0.f;
    for (int r = tid; r < M; r += 256) {
        float v = (float)A[(size_t)r * ld + c];
        a += v; b += v * v;
    }
    s1[tid] = a; s2[tid] = b; __syncthreads();
    for (int s = 128; s > 0; s >>= 1) {
        if (tid < s) { s1[tid] += s1[tid + s]; s2[tid] += s2[tid + s]; }
        __syncthreads();
    }
    if (tid == 0) {
        float mean = s1[0] / M;
        float var  = s2[0] / M - mean * mean;
        float sc = rsqrtf(var + eps);
        scale[c] = sc; shift[c] = -mean * sc;
    }
}

__device__ __forceinline__ void stats256(float v, float* sbuf, float eps,
                                         float* mean, float* rstd)
{
    int tid = threadIdx.x;
    __syncthreads();
    sbuf[tid] = v; sbuf[256 + tid] = v * v;
    __syncthreads();
    for (int s = 128; s > 0; s >>= 1) {
        if (tid < s) { sbuf[tid] += sbuf[tid + s]; sbuf[256 + tid] += sbuf[256 + tid + s]; }
        __syncthreads();
    }
    float m = sbuf[0] / 256.f;
    float var = sbuf[256] / 256.f - m * m;
    *mean = m; *rstd = rsqrtf(var + eps);
}

// LN -> +positional encoding -> LN (one block per token)
__global__ void ln_pe_kernel(const float* __restrict__ emb,
                             const float* g1, const float* b1,
                             const float* g2, const float* b2,
                             float* __restrict__ xf, bf16* __restrict__ xh)
{
    __shared__ float sbuf[512];
    int row = blockIdx.x, c = threadIdx.x;
    float v = emb[(size_t)row * 256 + c];
    float m, rs;
    stats256(v, sbuf, 1e-6f, &m, &rs);
    float y = g1[c] * (v - m) * rs + b1[c];
    int t = row % NTOK;
    float i2 = (float)(c & ~1);
    float ang = (float)t * __expf(-__logf(10000.f) * i2 / 256.f);
    y += (c & 1) ? __cosf(ang) : __sinf(ang);
    stats256(y, sbuf, 1e-6f, &m, &rs);
    float o = g2[c] * (y - m) * rs + b2[c];
    xf[(size_t)row * 256 + c] = o;
    xh[(size_t)row * 256 + c] = (bf16)o;
}

// attention: one block per (q, h, b); softmax over 140 keys, DK=DV=64
__global__ void attn_kernel(const float* __restrict__ Q, const float* __restrict__ Kf,
                            const float* __restrict__ V, const unsigned char* __restrict__ mask,
                            bf16* __restrict__ ctx)
{
    int qi = blockIdx.x, h = blockIdx.y, b = blockIdx.z;
    int tid = threadIdx.x;                 // 160 threads
    __shared__ float p[NTOK];
    __shared__ float sm;
    const float* qv = &Q[(size_t)(b * NTOK + qi) * 256 + h * 64];
    if (tid < NTOK) {
        const float* kv = &Kf[(size_t)(b * NTOK + tid) * 256 + h * 64];
        float d = 0.f;
        for (int e = 0; e < 64; ++e) d += qv[e] * kv[e];
        d *= 0.125f;
        if (mask[(size_t)(b * NTOK + qi) * NTOK + tid]) d = -1e9f;
        p[tid] = d;
    }
    __syncthreads();
    if (tid == 0) {
        float m = -1e30f;
        for (int k = 0; k < NTOK; ++k) m = fmaxf(m, p[k]);
        float su = 0.f;
        for (int k = 0; k < NTOK; ++k) { float e = __expf(p[k] - m); p[k] = e; su += e; }
        sm = su;
    }
    __syncthreads();
    if (tid < 64) {
        float inv = 1.f / sm;
        float acc = 0.f;
        for (int k = 0; k < NTOK; ++k)
            acc += p[k] * V[(size_t)(b * NTOK + k) * 256 + h * 64 + tid];
        ctx[(size_t)(b * NTOK + qi) * 256 + h * 64 + tid] = (bf16)(acc * inv);
    }
}

// cnn head: (already-normalized h3) @ fc4 + b   (N=3, linear)
__global__ void head3_kernel(const bf16* __restrict__ h3, const float* __restrict__ w,
                             const float* __restrict__ b, float* __restrict__ out)
{
    int row = blockIdx.x, tid = threadIdx.x;
    __shared__ float sv[256];
    sv[tid] = (float)h3[(size_t)row * 256 + tid];
    __syncthreads();
    if (tid < 3) {
        float s = b[tid];
        for (int k = 0; k < 256; ++k) s += sv[k] * w[k * 3 + tid];
        out[(size_t)row * 3 + tid] = s;
    }
}

// final head: sigmoid(x @ out_w + out_b)
__global__ void head_out_kernel(const float* __restrict__ xf, const float* __restrict__ w,
                                const float* __restrict__ b, float* __restrict__ out)
{
    int row = blockIdx.x, tid = threadIdx.x;
    __shared__ float sv[256];
    sv[tid] = xf[(size_t)row * 256 + tid];
    __syncthreads();
    if (tid < 3) {
        float s = b[tid];
        for (int k = 0; k < 256; ++k) s += sv[k] * w[k * 3 + tid];
        out[(size_t)row * 3 + tid] = 1.f / (1.f + __expf(-s));
    }
}

// ---------------------------------------------------------------------------
extern "C" void kernel_launch(void* const* d_in, const int* in_sizes, int n_in,
                              void* d_out, int out_size, void* d_ws, size_t ws_size,
                              hipStream_t stream)
{
    (void)in_sizes; (void)n_in; (void)out_size; (void)ws_size;
    const float* x      = (const float*)d_in[0];
    const unsigned char* amask = (const unsigned char*)d_in[2];
    const float* w1 = (const float*)d_in[3];  const float* b1 = (const float*)d_in[4];
    const float* w2 = (const float*)d_in[5];  const float* b2 = (const float*)d_in[6];
    const float* w3 = (const float*)d_in[7];  const float* b3 = (const float*)d_in[8];
    const float* w4 = (const float*)d_in[9];  const float* b4 = (const float*)d_in[10];
    const float* fc1_w = (const float*)d_in[11]; const float* fc1_b = (const float*)d_in[12];
    const float* fc2_w = (const float*)d_in[13]; const float* fc2_b = (const float*)d_in[14];
    const float* fc3_w = (const float*)d_in[15]; const float* fc3_b = (const float*)d_in[16];
    const float* fc4_w = (const float*)d_in[17]; const float* fc4_b = (const float*)d_in[18];
    const float* ln1_g = (const float*)d_in[19]; const float* ln1_b = (const float*)d_in[20];
    const float* ln2_g = (const float*)d_in[21]; const float* ln2_b = (const float*)d_in[22];
    const float* blk_wq = (const float*)d_in[23]; const float* blk_bq = (const float*)d_in[24];
    const float* blk_wk = (const float*)d_in[25]; const float* blk_bk = (const float*)d_in[26];
    const float* blk_wv = (const float*)d_in[27]; const float* blk_bv = (const float*)d_in[28];
    const float* blk_wo = (const float*)d_in[29]; const float* blk_bo = (const float*)d_in[30];
    const float* blk_ffw1 = (const float*)d_in[31]; const float* blk_ffb1 = (const float*)d_in[32];
    const float* blk_ffw2 = (const float*)d_in[33]; const float* blk_ffb2 = (const float*)d_in[34];
    const float* blk_ln1g = (const float*)d_in[35]; const float* blk_ln1b = (const float*)d_in[36];
    const float* blk_ln2g = (const float*)d_in[37]; const float* blk_ln2b = (const float*)d_in[38];
    const float* out_w = (const float*)d_in[39]; const float* out_b = (const float*)d_in[40];
    float* out = (float*)d_out;

    const size_t G = NGROUP;
    char* ws = (char*)d_ws;
    size_t off = 0;
    auto take = [&](size_t bytes) { size_t o = off; off += (bytes + 255) & ~(size_t)255; return o; };
    bf16*  NEWX = (bf16*) (ws + take(G * 128 * 2));
    bf16*  A4   = (bf16*) (ws + take(G * 1536 * 2));
    bf16*  A5   = (bf16*) (ws + take(G * 3008 * 2));      // ld 3008, pad cols zeroed
    float* SC1  = (float*)(ws + take(3008 * 4));
    float* SH1  = (float*)(ws + take(3008 * 4));
    float* EMB  = (float*)(ws + take(G * 256 * 4));
    bf16*  EMBR = (bf16*) (ws + take(G * 256 * 2));
    bf16*  H3   = (bf16*) (ws + take(G * 256 * 2));
    float* SC2  = (float*)(ws + take(256 * 4));
    float* SH2  = (float*)(ws + take(256 * 4));
    float* SC3  = (float*)(ws + take(256 * 4));
    float* SH3  = (float*)(ws + take(256 * 4));
    float* XAF  = (float*)(ws + take(G * 256 * 4));
    float* XBF  = (float*)(ws + take(G * 256 * 4));
    bf16*  XAH  = (bf16*) (ws + take(G * 256 * 2));
    bf16*  XBH  = (bf16*) (ws + take(G * 256 * 2));
    // pre-transposed bf16 weights [N x Kp]
    bf16*  WTw4  = (bf16*)(ws + take((size_t)1536 * 128 * 2));
    bf16*  WTfc1 = (bf16*)(ws + take((size_t)3008 * 1536 * 2));
    bf16*  WTfc2 = (bf16*)(ws + take((size_t)256 * 3008 * 2));
    bf16*  WTfc3 = (bf16*)(ws + take((size_t)256 * 256 * 2));
    const size_t PERBLK = 4 * 256 * 256 + (size_t)2048 * 256 + (size_t)256 * 2048;
    bf16*  WTBLK = (bf16*)(ws + take(4 * PERBLK * 2));
    // aliases into dead regions (A5 dead after fc2, A4 dead after fc1)
    float* Qf = (float*)A5;
    float* Kf = Qf + G * 256;
    float* Vf = Kf + G * 256;
    bf16*  CTX  = A4;
    bf16*  FBUF = (bf16*)A5;

    auto conv = [&](const float* W, int K, int N, int Kp, bf16* Wt) {
        dim3 grid(Kp / 32, (N + 31) / 32);
        wt_conv_kernel<<<grid, 256, 0, stream>>>(W, K, N, Kp, Wt);
    };
    auto gemm = [&](const bf16* A, int lda, const bf16* Wt, int Kp, int N, int Npad,
                    const float* bias, float* of, int ldcf, int reluf,
                    bf16* oh, int ldch, int reluh) {
        dim3 grid(NGROUP / 16, Npad / 64);
        gemm_bf16_kernel<<<grid, 32, 0, stream>>>(A, lda, Wt, Kp, N, bias,
                                                  of, ldcf, reluf, oh, ldch, reluh);
    };

    // ---- one-time (per launch) weight transpose/convert to bf16 ----
    conv(w4,    99,   1536, 128,  WTw4);
    conv(fc1_w, 1536, 3000, 1536, WTfc1);
    conv(fc2_w, 3000, 256,  3008, WTfc2);
    conv(fc3_w, 256,  256,  256,  WTfc3);
    bf16* WTq[4]; bf16* WTk[4]; bf16* WTv[4]; bf16* WTo[4]; bf16* WTf1[4]; bf16* WTf2[4];
    for (int i = 0; i < 4; ++i) {
        bf16* base = WTBLK + (size_t)i * PERBLK;
        WTq[i]  = base;                 WTk[i]  = WTq[i]  + 256 * 256;
        WTv[i]  = WTk[i]  + 256 * 256;  WTo[i]  = WTv[i]  + 256 * 256;
        WTf1[i] = WTo[i]  + 256 * 256;  WTf2[i] = WTf1[i] + (size_t)2048 * 256;
        conv(blk_wq  + (size_t)i * 256 * 256,  256,  256,  256,  WTq[i]);
        conv(blk_wk  + (size_t)i * 256 * 256,  256,  256,  256,  WTk[i]);
        conv(blk_wv  + (size_t)i * 256 * 256,  256,  256,  256,  WTv[i]);
        conv(blk_wo  + (size_t)i * 256 * 256,  256,  256,  256,  WTo[i]);
        conv(blk_ffw1 + (size_t)i * 256 * 2048, 256,  2048, 256,  WTf1[i]);
        conv(blk_ffw2 + (size_t)i * 2048 * 256, 2048, 256,  2048, WTf2[i]);
    }

    // ---- stage 1: frame MLP + chunk pooling ----
    frame_mlp_kernel<<<NGROUP, 128, 0, stream>>>(x, w1, b1, w2, b2, w3, b3, NEWX);

    // ---- x-vector trunk ----
    gemm(NEWX, 128, WTw4, 128, 1536, 1536, b4, nullptr, 0, 0, A4, 1536, 1);
    gemm(A4, 1536, WTfc1, 1536, 3000, 3008, fc1_b, nullptr, 0, 0, A5, 3008, 1);
    col_stats_kernel<<<3000, 256, 0, stream>>>(A5, NGROUP, 3008, 1e-5f, SC1, SH1);
    bn_apply_kernel<<<(int)((G * 3000 + 255) / 256), 256, 0, stream>>>(A5, NGROUP, 3000, 3008, SC1, SH1);
    gemm(A5, 3008, WTfc2, 3008, 256, 256, fc2_b, EMB, 256, 0, EMBR, 256, 1);
    col_stats_kernel<<<256, 256, 0, stream>>>(EMBR, NGROUP, 256, 1e-5f, SC2, SH2);
    bn_apply_kernel<<<(int)((G * 256 + 255) / 256), 256, 0, stream>>>(EMBR, NGROUP, 256, 256, SC2, SH2);
    gemm(EMBR, 256, WTfc3, 256, 256, 256, fc3_b, nullptr, 0, 0, H3, 256, 1);
    col_stats_kernel<<<256, 256, 0, stream>>>(H3, NGROUP, 256, 1e-5f, SC3, SH3);
    bn_apply_kernel<<<(int)((G * 256 + 255) / 256), 256, 0, stream>>>(H3, NGROUP, 256, 256, SC3, SH3);
    head3_kernel<<<NGROUP, 256, 0, stream>>>(H3, fc4_w, fc4_b, out + (size_t)NGROUP * 3);

    // ---- transformer encoder ----
    ln_pe_kernel<<<NGROUP, 256, 0, stream>>>(EMB, ln1_g, ln1_b, ln2_g, ln2_b, XAF, XAH);
    float* curF = XAF; bf16* curH = XAH;
    float* midF = XBF; bf16* midH = XBH;
    for (int i = 0; i < 4; ++i) {
        gemm(curH, 256, WTq[i], 256, 256, 256, blk_bq + i * 256, Qf, 256, 0, nullptr, 0, 0);
        gemm(curH, 256, WTk[i], 256, 256, 256, blk_bk + i * 256, Kf, 256, 0, nullptr, 0, 0);
        gemm(curH, 256, WTv[i], 256, 256, 256, blk_bv + i * 256, Vf, 256, 0, nullptr, 0, 0);
        attn_kernel<<<dim3(NTOK, 4, NBATCH), 160, 0, stream>>>(Qf, Kf, Vf, amask, CTX);
        gemm_ln_kernel<<<NGROUP / 16, 256, 0, stream>>>(CTX, 256, 256, WTo[i], blk_bo + i * 256,
                                                        curF, blk_ln1g + i * 256, blk_ln1b + i * 256,
                                                        midF, midH);
        gemm(midH, 256, WTf1[i], 256, 2048, 2048, blk_ffb1 + i * 2048,
             nullptr, 0, 0, FBUF, 2048, 1);
        gemm_ln_kernel<<<NGROUP / 16, 256, 0, stream>>>(FBUF, 2048, 2048, WTf2[i],
                                                        blk_ffb2 + i * 256, midF,
                                                        blk_ln2g + i * 256, blk_ln2b + i * 256,
                                                        curF, curH);
    }
    head_out_kernel<<<NGROUP, 256, 0, stream>>>(curF, out_w, out_b, out);
}
